// NoisyTopKRouter_11029476016644
// MI455X (gfx1250) — matrix operand order repaired and checked by reading, not compiled, not run on hardware
//
#include <hip/hip_runtime.h>
#include <hip/hip_bf16.h>

// B=4, S=4096, D=2048, E=64, top_k=8
// out  = softmax over top-8 of (x @ Wn + bn), zeros elsewhere   [B,S,E] f32
// idx  = top-8 indices (descending value, lowest index on ties) [B,S,8] i32
// (Wr, br, noise are dead code for the outputs.)

typedef __bf16 bf16;
typedef __attribute__((ext_vector_type(16))) __bf16 v16bf;
typedef __attribute__((ext_vector_type(8)))  float  v8f;
typedef __attribute__((ext_vector_type(4)))  float  v4f;
typedef __attribute__((ext_vector_type(4)))  int    v4i;

#define DDIM 2048
#define EDIM 64
#define TOPK 8
#define ROWS_TOTAL 16384        /* B*S */
#define KCHUNKS (DDIM / 32)     /* 64  */
#define LDS_STRIDE 68           /* 64 + 4 pad: bank-spread, rows 16B-aligned */

// ---------------------------------------------------------------------------
// Kernel 1: swizzle Wn (f32 [2048][64]) into bf16 B-fragments matching the
// CDNA5 16-bit B 32x16 WMMA layout: frag index f = (kc*4 + j)*32 + lane,
// lane<16  -> column N = 16j+lane,     K = kc*32 + {0..7, 16..23}
// lane>=16 -> column N = 16j+lane-16,  K = kc*32 + {8..15, 24..31}
// Each fragment lane holds 16 bf16 = 32 contiguous bytes -> b128 loads later.
// ---------------------------------------------------------------------------
__global__ __launch_bounds__(256) void swizzle_wn_kernel(
    const float* __restrict__ Wn, bf16* __restrict__ wsB) {
  int t = blockIdx.x * blockDim.x + threadIdx.x;   // 0 .. KCHUNKS*4*32 - 1
  if (t >= KCHUNKS * 4 * 32) return;
  int lane = t & 31;
  int j    = (t >> 5) & 3;
  int kc   = t >> 7;
  int n    = lane & 15;
  int k0   = kc * 32 + ((lane >> 4) ? 8 : 0);
  int N    = j * 16 + n;
  v16bf frag;
#pragma unroll
  for (int e = 0; e < 16; ++e)
    frag[e] = (bf16)Wn[(size_t)(k0 + e + (e >= 8 ? 8 : 0)) * EDIM + N];
  *(v16bf*)(wsB + (size_t)t * 16) = frag;
}

// ---------------------------------------------------------------------------
// Kernel 2: GEMM (bf16 WMMA, f32 accum) + bias + top-8 + softmax scatter.
// 256 threads = 8 waves; wave w owns rows [blockRow + 16w, +16), all 64 cols.
// ---------------------------------------------------------------------------
__global__ __launch_bounds__(256) void router_topk_kernel(
    const float* __restrict__ x,
    const float* __restrict__ bn,
    const bf16*  __restrict__ wsB,
    float* __restrict__ outP,
    int*   __restrict__ idxP) {
  __shared__ float tile[128 * LDS_STRIDE];   // 34,816 B

  const int tid  = threadIdx.x;
  const int wave = tid >> 5;
  const int lane = tid & 31;
  const int l16  = lane & 15;
  const int hi   = lane >> 4;
  const int blockRow = blockIdx.x * 128;
  const size_t row = (size_t)blockRow + wave * 16 + l16;
  const int khalf = hi * 8;   // 16-bit A layout: lanes 0-15 K{0-7,16-23}, 16-31 K{8-15,24-31}

  v8f acc0 = {}, acc1 = {}, acc2 = {}, acc3 = {};
  const float* xr = x + row * DDIM;
  const v16bf* bfrag = (const v16bf*)wsB;

  for (int kc = 0; kc < KCHUNKS; ++kc) {
    const int base = kc * 32 + khalf;
    v4f f0 = *(const v4f*)(xr + base);
    v4f f1 = *(const v4f*)(xr + base + 4);
    v4f f2 = *(const v4f*)(xr + base + 16);
    v4f f3 = *(const v4f*)(xr + base + 20);
    v16bf a;
#pragma unroll
    for (int e = 0; e < 4; ++e) {
      a[e]      = (bf16)f0[e];
      a[4 + e]  = (bf16)f1[e];
      a[8 + e]  = (bf16)f2[e];
      a[12 + e] = (bf16)f3[e];
    }
    const v16bf* bq = bfrag + (size_t)(kc * 4) * 32 + lane;
    v16bf b0 = bq[0];        // j=0 fragment (32B -> 2x global_load_b128)
    v16bf b1 = bq[32];       // j=1
    v16bf b2 = bq[64];       // j=2
    v16bf b3 = bq[96];       // j=3
    acc0 = __builtin_amdgcn_wmma_f32_16x16x32_bf16(false, a, false, b0, (short)0, acc0, false, false);
    acc1 = __builtin_amdgcn_wmma_f32_16x16x32_bf16(false, a, false, b1, (short)0, acc1, false, false);
    acc2 = __builtin_amdgcn_wmma_f32_16x16x32_bf16(false, a, false, b2, (short)0, acc2, false, false);
    acc3 = __builtin_amdgcn_wmma_f32_16x16x32_bf16(false, a, false, b3, (short)0, acc3, false, false);
  }

  // C/D layout: lane L -> N = (L&15) + 16j, VGPR v -> M = v + 8*(L>>4).
  float bb0 = bn[l16], bb1 = bn[16 + l16], bb2 = bn[32 + l16], bb3 = bn[48 + l16];
  const int rbase = wave * 16 + hi * 8;
#pragma unroll
  for (int v = 0; v < 8; ++v) {
    float* rp = &tile[(rbase + v) * LDS_STRIDE + l16];
    rp[0]  = acc0[v] + bb0;
    rp[16] = acc1[v] + bb1;
    rp[32] = acc2[v] + bb2;
    rp[48] = acc3[v] + bb3;
  }
  __syncthreads();

  // Per-row top-8 (strict '>' => lowest index wins ties, matching lax.top_k),
  // softmax over the 8 kept values, scatter into zeroed row.
  if (tid < 128) {
    float* rowp = &tile[tid * LDS_STRIDE];
    float vals[TOPK];
    int   idxs[TOPK];
#pragma unroll 1
    for (int k = 0; k < TOPK; ++k) {
      float best = -__builtin_inff();
      int bi = 0;
      for (int c = 0; c < EDIM; ++c) {
        float v = rowp[c];
        if (v > best) { best = v; bi = c; }
      }
      vals[k] = best;
      idxs[k] = bi;
      rowp[bi] = -__builtin_inff();
    }
    float m = vals[0];   // vals[0] is the max
    float s = 0.f, p[TOPK];
#pragma unroll
    for (int k = 0; k < TOPK; ++k) { p[k] = __expf(vals[k] - m); s += p[k]; }
    float inv = 1.f / s;
    for (int c = 0; c < EDIM; ++c) rowp[c] = 0.f;
#pragma unroll
    for (int k = 0; k < TOPK; ++k) rowp[idxs[k]] = p[k] * inv;

    v4i i0 = { idxs[0], idxs[1], idxs[2], idxs[3] };
    v4i i1 = { idxs[4], idxs[5], idxs[6], idxs[7] };
    int* ip = idxP + (size_t)(blockRow + tid) * TOPK;
    *(v4i*)ip       = i0;
    *(v4i*)(ip + 4) = i1;
  }
  __syncthreads();

  // Coalesced float4 store of the 128x64 probability tile.
  for (int i = tid; i < 128 * (EDIM / 4); i += 256) {
    int r = i >> 4;
    int q = i & 15;
    v4f v = *(v4f*)&tile[r * LDS_STRIDE + q * 4];
    *(v4f*)&outP[(size_t)(blockRow + r) * EDIM + q * 4] = v;
  }
}

// ---------------------------------------------------------------------------
extern "C" void kernel_launch(void* const* d_in, const int* in_sizes, int n_in,
                              void* d_out, int out_size, void* d_ws, size_t ws_size,
                              hipStream_t stream) {
  // setup_inputs order: 0:x 1:Wr 2:br 3:Wn 4:bn 5:noise 6:top_k
  const float* x  = (const float*)d_in[0];
  const float* Wn = (const float*)d_in[3];
  const float* bn = (const float*)d_in[4];

  float* outP = (float*)d_out;                          // [16384, 64] f32
  int*   idxP = (int*)d_out + (size_t)ROWS_TOTAL * EDIM; // [16384, 8] i32 (tuple tail)

  bf16* wsB = (bf16*)d_ws;  // needs 64*4*32*16*2 = 256 KiB of scratch

  const int nfrag_threads = KCHUNKS * 4 * 32;           // 8192
  swizzle_wn_kernel<<<(nfrag_threads + 255) / 256, 256, 0, stream>>>(Wn, wsB);
  router_topk_kernel<<<ROWS_TOTAL / 128, 256, 0, stream>>>(x, bn, wsB, outP, idxP);
}
